// LSTM_76656576299174
// MI455X (gfx1250) — compile-verified, matrix operand order checked by reading, not run on hardware
//
#include <hip/hip_runtime.h>

typedef __attribute__((ext_vector_type(16))) _Float16 v16h;
typedef __attribute__((ext_vector_type(8)))  _Float16 v8h;
typedef __attribute__((ext_vector_type(8)))  float    v8f;

#define T_LEN   65536
#define INP     6
#define H       100
#define G4      400     // real gate rows (4*H)
#define GP      512     // padded gate rows -> 32 row tiles, uniform across 16 waves
#define NRT     32      // padded row tiles (tiles 25..31 are zero-weight dummies)
#define NRT_REAL 25
#define KT0     4       // layer0 K = 128 : [x(6) | h0(100) | pad(22)]
#define KT1     7       // layer1 K = 224 : [h1(100) | pad(12) | h2(100) | pad(12)]
#define NW      16      // waves per block (wave32)
#define THREADS 512
#define SLOTS   2       // row-tile slots per wave: 16 waves * 2 = 32 tiles

// ---- fast transcendentals on the serial critical path (no IEEE divides) ----
__device__ __forceinline__ float fast_tanh(float x) {
#if __has_builtin(__builtin_amdgcn_tanhf)
  return __builtin_amdgcn_tanhf(x);       // gfx1250 v_tanh_f32
#else
  x = fminf(fmaxf(x, -15.f), 15.f);
  float e = __expf(2.f * x);
  return 1.f - 2.f * __builtin_amdgcn_rcpf(e + 1.f);
#endif
}
__device__ __forceinline__ float fast_sigm(float x) {
  return 0.5f + 0.5f * fast_tanh(0.5f * x);
}

__device__ __forceinline__ v8f wmma16(v16h a, v16h b, v8f c) {
  return __builtin_amdgcn_wmma_f32_16x16x32_f16(false, a, false, b,
                                                (short)0, c, false, false);
}

__global__ void __launch_bounds__(THREADS, 1)
lstm_wmma_kernel(const float* __restrict__ x,
                 const float* __restrict__ w_ih0, const float* __restrict__ w_hh0,
                 const float* __restrict__ b_ih0, const float* __restrict__ b_hh0,
                 const float* __restrict__ w_ih1, const float* __restrict__ w_hh1,
                 const float* __restrict__ b_ih1, const float* __restrict__ b_hh1,
                 const float* __restrict__ w_lin, const float* __restrict__ b_lin,
                 float* __restrict__ out) {
  // small LDS only: gates, biases, state vectors (~8 KB)
  __shared__ float bs0[G4], bs1[G4], g[GP];
  __shared__ alignas(16) _Float16 u0[128];   // [x(6) | h0(100) | pad]
  __shared__ alignas(16) _Float16 u1[224];   // [h1(100) | pad | h2(100) | pad]
  __shared__ float c1[H], c2[H], h2f[H];

  const int tid  = threadIdx.x;
  const int lane = tid & 31;          // wave32
  const int wave = tid >> 5;          // 0..15
  const int m    = lane & 15;         // row within 16x32 A tile
  const int bk   = (lane >> 4) << 3;  // K base: 0 (lanes 0-15) / 8 (lanes 16-31)

  // -------- register-resident weight A-fragments (f16), loaded once --------
  // 2 slots * (4+7) K-tiles * 8 VGPRs = 176 VGPRs/wave for weights.
  // A layout (ISA 7.12.2): lane<16 holds M=lane, K={bk..bk+7, bk+16..bk+23}
  v16h a0[SLOTS][KT0];
  v16h a1[SLOTS][KT1];
#pragma unroll
  for (int s = 0; s < SLOTS; ++s) {
    const int  rt    = wave + NW * s;        // 0..31, uniform per wave
    const int  row   = rt * 16 + m;
    const bool valid = (rt < NRT_REAL);      // dummy tiles get zero weights
#pragma unroll
    for (int kt = 0; kt < KT0; ++kt) {
#pragma unroll
      for (int i = 0; i < 16; ++i) {
        const int k = kt * 32 + bk + i + (i >= 8 ? 8 : 0);
        float v = 0.f;
        if (valid) {
          if (k < INP)          v = w_ih0[row * INP + k];
          else if (k < INP + H) v = w_hh0[row * H + (k - INP)];
        }
        a0[s][kt][i] = (_Float16)v;
      }
    }
#pragma unroll
    for (int kt = 0; kt < KT1; ++kt) {
#pragma unroll
      for (int i = 0; i < 16; ++i) {
        const int k = kt * 32 + bk + i + (i >= 8 ? 8 : 0);
        float v = 0.f;
        if (valid) {
          if (k < H)                    v = w_ih1[row * H + k];
          else if (k >= 112 && k < 212) v = w_hh1[row * H + (k - 112)];
        }
        a1[s][kt][i] = (_Float16)v;
      }
    }
  }

  // -------- init LDS state --------
  for (int i = tid; i < G4; i += THREADS) {
    bs0[i] = b_ih0[i] + b_hh0[i];
    bs1[i] = b_ih1[i] + b_hh1[i];
  }
  for (int i = tid; i < 128; i += THREADS) u0[i] = (_Float16)0.f;
  for (int i = tid; i < 224; i += THREADS) u1[i] = (_Float16)0.f;
  for (int i = tid; i < H; i += THREADS) { c1[i] = 0.f; c2[i] = 0.f; h2f[i] = 0.f; }
  __syncthreads();

  float xr = (tid < INP) ? x[tid] : 0.f;   // prefetched input

  // -------- 65536 dependent steps --------
#pragma unroll 1
  for (int t = 0; t < T_LEN; ++t) {
    if (tid < INP) u0[tid] = (_Float16)xr;
    __syncthreads();
    if (tid < INP) xr = x[((t + 1 < T_LEN) ? t + 1 : t) * INP + tid];  // prefetch

    // ---- layer 0: g = [Wih0|Whh0] @ [x ; h0] ---- (uniform, branch-free)
    {
      v8f acc[SLOTS] = {};
#pragma unroll
      for (int kt = 0; kt < KT0; ++kt) {
        // B fragment: all 16 columns replicated = u segment (mirror-of-A K map)
        v8h lo = *(const v8h*)(u0 + kt * 32 + bk);
        v8h hi = *(const v8h*)(u0 + kt * 32 + bk + 16);
        v16h b;
#pragma unroll
        for (int i = 0; i < 8; ++i) { b[i] = lo[i]; b[i + 8] = hi[i]; }
#pragma unroll
        for (int s = 0; s < SLOTS; ++s)          // 2 independent WMMA chains
          acc[s] = wmma16(a0[s][kt], b, acc[s]);
      }
#pragma unroll
      for (int s = 0; s < SLOTS; ++s) {
        if (m == 0) {                            // lane0: M 0..7, lane16: M 8..15
          const int mb = (wave + NW * s) * 16 + bk;
#pragma unroll
          for (int i = 0; i < 8; ++i) g[mb + i] = acc[s][i];
        }
      }
    }
    __syncthreads();

    if (tid < H) {
      float gi = fast_sigm(g[tid]       + bs0[tid]);
      float gf = fast_sigm(g[tid + 100] + bs0[tid + 100]);
      float gg = fast_tanh(g[tid + 200] + bs0[tid + 200]);
      float go = fast_sigm(g[tid + 300] + bs0[tid + 300]);
      float c  = gf * c1[tid] + gi * gg;
      c1[tid] = c;
      float h = go * fast_tanh(c);
      u0[INP + tid] = (_Float16)h;   // h0 for next step
      u1[tid]       = (_Float16)h;   // input to layer 1
    }
    __syncthreads();

    // ---- layer 1: g = [Wih1|Whh1] @ [h1 ; h2_prev] ----
    {
      v8f acc[SLOTS] = {};
#pragma unroll
      for (int kt = 0; kt < KT1; ++kt) {
        v8h lo = *(const v8h*)(u1 + kt * 32 + bk);
        v8h hi = *(const v8h*)(u1 + kt * 32 + bk + 16);
        v16h b;
#pragma unroll
        for (int i = 0; i < 8; ++i) { b[i] = lo[i]; b[i + 8] = hi[i]; }
#pragma unroll
        for (int s = 0; s < SLOTS; ++s)
          acc[s] = wmma16(a1[s][kt], b, acc[s]);
      }
#pragma unroll
      for (int s = 0; s < SLOTS; ++s) {
        if (m == 0) {
          const int mb = (wave + NW * s) * 16 + bk;
#pragma unroll
          for (int i = 0; i < 8; ++i) g[mb + i] = acc[s][i];
        }
      }
    }
    __syncthreads();

    if (tid < H) {
      float gi = fast_sigm(g[tid]       + bs1[tid]);
      float gf = fast_sigm(g[tid + 100] + bs1[tid + 100]);
      float gg = fast_tanh(g[tid + 200] + bs1[tid + 200]);
      float go = fast_sigm(g[tid + 300] + bs1[tid + 300]);
      float c  = gf * c2[tid] + gi * gg;
      c2[tid] = c;
      float h = go * fast_tanh(c);
      u1[112 + tid] = (_Float16)h;   // h2 for next step
      h2f[tid] = h;
    }
    __syncthreads();
  }

  // final linear on last hidden state
  if (tid == 0) {
    float s = b_lin[0];
    for (int j = 0; j < H; ++j) s += w_lin[j] * h2f[j];
    out[0] = s;
  }
}

extern "C" void kernel_launch(void* const* d_in, const int* in_sizes, int n_in,
                              void* d_out, int out_size, void* d_ws, size_t ws_size,
                              hipStream_t stream) {
  (void)in_sizes; (void)n_in; (void)out_size; (void)d_ws; (void)ws_size;
  const float* x     = (const float*)d_in[0];
  const float* w_ih0 = (const float*)d_in[1];
  const float* w_hh0 = (const float*)d_in[2];
  const float* b_ih0 = (const float*)d_in[3];
  const float* b_hh0 = (const float*)d_in[4];
  const float* w_ih1 = (const float*)d_in[5];
  const float* w_hh1 = (const float*)d_in[6];
  const float* b_ih1 = (const float*)d_in[7];
  const float* b_hh1 = (const float*)d_in[8];
  const float* w_lin = (const float*)d_in[9];
  const float* b_lin = (const float*)d_in[10];
  float* out = (float*)d_out;

  lstm_wmma_kernel<<<1, THREADS, 0, stream>>>(
      x, w_ih0, w_hh0, b_ih0, b_hh0,
      w_ih1, w_hh1, b_ih1, b_hh1, w_lin, b_lin, out);
}
